// multipole_conv2d_v1_3762391352102
// MI455X (gfx1250) — compile-verified, matrix operand order checked by reading, not run on hardware
//
#include <hip/hip_runtime.h>
#include <hip/hip_bf16.h>
#include <stddef.h>

// CDNA5 / gfx1250 WMMA types (wave32)
typedef __attribute__((ext_vector_type(16))) __bf16 v16bf;
typedef __attribute__((ext_vector_type(8)))  float  v8f;

static __device__ __forceinline__ v8f wmma_bf16(v16bf a, v16bf b, v8f c) {
  return __builtin_amdgcn_wmma_f32_16x16x32_bf16(false, a, false, b,
                                                 (short)0, c, false, false);
}

// ---------------------------------------------------------------------------
// Stage 1a: level 0: NCHW f32 -> NHWC bf16 via LDS tile transpose (coalesced
// both directions). Block = 256 threads handles a 64ch x 64w tile at fixed
// (b, h). Padded LDS row (65) keeps both phases bank-conflict free.
// ---------------------------------------------------------------------------
__global__ __launch_bounds__(256)
void mpole_x_to_nhwc(const float* __restrict__ x, __bf16* __restrict__ out) {
  __shared__ float tile[64][65];
  const int H = 256, W = 256;
  int blk = blockIdx.x;                 // 4 * 256 * (256/64) = 4096 blocks
  int w0  = (blk & 3) << 6;
  int t   = blk >> 2;
  int h   = t & 255;
  int b   = t >> 8;

  int tw  = threadIdx.x & 63;           // w within tile (coalesced read dim)
  int tc4 = threadIdx.x >> 6;           // 0..3
#pragma unroll
  for (int i = 0; i < 16; ++i) {
    int c = tc4 * 16 + i;
    tile[c][tw] = x[((size_t)(b * 64 + c) * H + h) * W + w0 + tw];
  }
  __syncthreads();

  int c  = threadIdx.x & 63;            // channel (coalesced write dim)
  int pg = threadIdx.x >> 6;            // pixel subgroup 0..3
  size_t rowbase = ((size_t)(b * H + h) * W + w0) * 64;
#pragma unroll
  for (int i = 0; i < 16; ++i) {
    int w = 4 * i + pg;
    out[rowbase + (size_t)w * 64 + c] = (__bf16)tile[c][w];
  }
}

// ---------------------------------------------------------------------------
// Stage 1b: pooled levels 1..3: NCHW f32 -> NHWC bf16, averaging s x s blocks.
// Small traffic (<11 MB total), simple per-element mapping is fine.
// ---------------------------------------------------------------------------
__global__ void mpole_pool_nhwc_bf16(const float* __restrict__ x,
                                     __bf16* __restrict__ out,
                                     int Hl, int Wl, int s) {
  int idx = blockIdx.x * blockDim.x + threadIdx.x;
  int total = 4 * Hl * Wl * 64;
  if (idx >= total) return;
  int c  = idx & 63;
  int t  = idx >> 6;
  int wo = t % Wl; t /= Wl;
  int ho = t % Hl;
  int b  = t / Hl;
  const int H = 256, W = 256;
  const float* xc = x + ((size_t)(b * 64 + c) * H) * W;
  float sum = 0.0f;
  for (int i = 0; i < s; ++i) {
    const float* row = xc + (size_t)(ho * s + i) * W + wo * s;
    for (int j = 0; j < s; ++j) sum += row[j];
  }
  out[idx] = (__bf16)(sum / (float)(s * s));
}

// ---------------------------------------------------------------------------
// Stage 2: weight reorder f32 -> bf16, k = tap-major.
// ---------------------------------------------------------------------------
__global__ void mpole_reorder_wlvl(const float* __restrict__ Ws,
                                   __bf16* __restrict__ out) {
  int idx = blockIdx.x * blockDim.x + threadIdx.x;   // 4*64*9*64 = 147456
  if (idx >= 4 * 64 * 9 * 64) return;
  int ci  = idx & 63;
  int t   = idx >> 6;
  int tap = t % 9; t /= 9;
  int o   = t & 63;
  int lvl = t >> 6;
  out[(size_t)(lvl * 64 + o) * 576 + tap * 64 + ci] =
      (__bf16)Ws[((size_t)((lvl * 64 + o) * 64 + ci)) * 9 + tap];
}

__global__ void mpole_reorder_wfin(const float* __restrict__ Wc,
                                   __bf16* __restrict__ out) {
  int idx = blockIdx.x * blockDim.x + threadIdx.x;   // 64*9*256 = 147456
  if (idx >= 64 * 9 * 256) return;
  int ch  = idx & 255;
  int t   = idx >> 8;
  int tap = t % 9;
  int o   = t / 9;
  out[(size_t)o * 2304 + tap * 256 + ch] =
      (__bf16)Wc[((size_t)(o * 256 + ch)) * 9 + tap];
}

// ---------------------------------------------------------------------------
// Stage 3: per-level 3x3 circular conv (64 -> 64) at level resolution,
// bias + nearest-upsample write into NHWC bf16 cat buffer.
// Block = 128 = 4 waves; each wave owns a DIFFERENT 16-pixel tile and all 64
// out-channels via 4 independent accumulators (hides the 5-slot bf16
// WMMA->WMMA RAW hazard and reuses each A fragment 4x).
// All 9 wrapped tap base pointers are hoisted; tap/chunk loops fully unrolled
// so fragment loads are immediate-offset global_load_b128 pairs.
// ---------------------------------------------------------------------------
__global__ __launch_bounds__(128)
void mpole_conv_level(const __bf16* __restrict__ in,   // NHWC bf16 (4,Hi,Wi,64)
                      const __bf16* __restrict__ wt,   // [64][576] bf16
                      const float*  __restrict__ bias, // [64]
                      __bf16* __restrict__ cat,        // NHWC bf16 (4,256,256,256)
                      int Hi, int Wi, int lvl) {
  const int lane  = threadIdx.x & 31;
  const int wave  = threadIdx.x >> 5;
  const int col   = lane & 15;
  const int khalf = lane >> 4;

  int p0  = blockIdx.x * 64 + wave * 16;     // 16-aligned, Wi multiple of 16
  int w0  = p0 % Wi;
  int rem = p0 / Wi;
  int h   = rem % Hi;
  int b   = rem / Hi;
  int wpx = w0 + col;                        // this lane's A-row pixel

  // Wrapped coordinates (circular padding), hoisted out of the K loop.
  int hm = h - 1;   hm += (hm < 0) ? Hi : 0;
  int hp = h + 1;   hp -= (hp >= Hi) ? Hi : 0;
  int wm = wpx - 1; wm += (wm < 0) ? Wi : 0;
  int wp = wpx + 1; wp -= (wp >= Wi) ? Wi : 0;
  int hrow[3] = {hm, h, hp};
  int wcol[3] = {wm, wpx, wp};

  const __bf16* abase[9];
#pragma unroll
  for (int th = 0; th < 3; ++th)
#pragma unroll
    for (int tw = 0; tw < 3; ++tw)
      abase[th * 3 + tw] =
          in + (((size_t)(b * Hi + hrow[th]) * Wi + wcol[tw]) << 6) + khalf * 16;

  const __bf16* wb0 = wt + (size_t)(0 * 16 + col) * 576 + khalf * 16;
  const __bf16* wb1 = wt + (size_t)(1 * 16 + col) * 576 + khalf * 16;
  const __bf16* wb2 = wt + (size_t)(2 * 16 + col) * 576 + khalf * 16;
  const __bf16* wb3 = wt + (size_t)(3 * 16 + col) * 576 + khalf * 16;

  v8f acc0 = {}, acc1 = {}, acc2 = {}, acc3 = {};
#pragma unroll
  for (int tap = 0; tap < 9; ++tap) {
#pragma unroll
    for (int ch = 0; ch < 2; ++ch) {
      int ko = tap * 64 + ch * 32;
      v16bf a  = *(const v16bf*)(abase[tap] + ch * 32);
      v16bf b0 = *(const v16bf*)(wb0 + ko);
      v16bf b1 = *(const v16bf*)(wb1 + ko);
      v16bf b2 = *(const v16bf*)(wb2 + ko);
      v16bf b3 = *(const v16bf*)(wb3 + ko);
      acc0 = wmma_bf16(a, b0, acc0);
      acc1 = wmma_bf16(a, b1, acc1);
      acc2 = wmma_bf16(a, b2, acc2);
      acc3 = wmma_bf16(a, b3, acc3);
    }
  }

  // Epilogue: D layout -> lane holds N = col, M = r + 8*khalf per VGPR r.
  int s = 1 << lvl;
#define MPOLE_LVL_EPI(ACC, N)                                                   \
  {                                                                             \
    int   oc  = (N) * 16 + col;                                                 \
    float bv  = bias[oc];                                                       \
    int   cch = lvl * 64 + oc;                                                  \
    _Pragma("unroll")                                                           \
    for (int r = 0; r < 8; ++r) {                                               \
      int mm = r + khalf * 8;                                                   \
      int wq = w0 + mm;                                                         \
      __bf16 v = (__bf16)(ACC[r] + bv);                                         \
      for (int dy = 0; dy < s; ++dy) {                                          \
        size_t rb = ((size_t)(b * 256 + h * s + dy) * 256 + (size_t)wq * s) * 256; \
        for (int dx = 0; dx < s; ++dx) cat[rb + (size_t)dx * 256 + cch] = v;    \
      }                                                                         \
    }                                                                           \
  }
  MPOLE_LVL_EPI(acc0, 0)
  MPOLE_LVL_EPI(acc1, 1)
  MPOLE_LVL_EPI(acc2, 2)
  MPOLE_LVL_EPI(acc3, 3)
#undef MPOLE_LVL_EPI
}

// ---------------------------------------------------------------------------
// Stage 4: final 3x3 circular conv (256 -> 64). K = 2304 -> 72 chunks, 4 acc
// per wave -> 288 WMMAs per wave. cat (134 MB) is L2-resident (192 MB L2) so
// fragments stream from L2. Output NCHW f32 (8 contiguous floats per lane).
// ---------------------------------------------------------------------------
__global__ __launch_bounds__(128)
void mpole_conv_final(const __bf16* __restrict__ cat,  // NHWC bf16 (4,256,256,256)
                      const __bf16* __restrict__ wt,   // [64][2304] bf16
                      const float*  __restrict__ bc,   // [64]
                      float* __restrict__ out) {       // NCHW f32 (4,64,256,256)
  const int H = 256, W = 256;
  const int lane  = threadIdx.x & 31;
  const int wave  = threadIdx.x >> 5;
  const int col   = lane & 15;
  const int khalf = lane >> 4;

  int p0  = blockIdx.x * 64 + wave * 16;
  int w0  = p0 & 255;
  int rem = p0 >> 8;
  int h   = rem & 255;
  int b   = rem >> 8;
  int wpx = w0 + col;

  int hm = h - 1;   hm += (hm < 0) ? H : 0;
  int hp = h + 1;   hp -= (hp >= H) ? H : 0;
  int wm = wpx - 1; wm += (wm < 0) ? W : 0;
  int wp = wpx + 1; wp -= (wp >= W) ? W : 0;
  int hrow[3] = {hm, h, hp};
  int wcol[3] = {wm, wpx, wp};

  const __bf16* abase[9];
#pragma unroll
  for (int th = 0; th < 3; ++th)
#pragma unroll
    for (int tw = 0; tw < 3; ++tw)
      abase[th * 3 + tw] =
          cat + (((size_t)(b * H + hrow[th]) * W + wcol[tw]) << 8) + khalf * 16;

  const __bf16* wb0 = wt + (size_t)(0 * 16 + col) * 2304 + khalf * 16;
  const __bf16* wb1 = wt + (size_t)(1 * 16 + col) * 2304 + khalf * 16;
  const __bf16* wb2 = wt + (size_t)(2 * 16 + col) * 2304 + khalf * 16;
  const __bf16* wb3 = wt + (size_t)(3 * 16 + col) * 2304 + khalf * 16;

  v8f acc0 = {}, acc1 = {}, acc2 = {}, acc3 = {};
#pragma unroll
  for (int tap = 0; tap < 9; ++tap) {
#pragma unroll
    for (int ch = 0; ch < 8; ++ch) {
      int ko = tap * 256 + ch * 32;
      v16bf a  = *(const v16bf*)(abase[tap] + ch * 32);
      v16bf b0 = *(const v16bf*)(wb0 + ko);
      v16bf b1 = *(const v16bf*)(wb1 + ko);
      v16bf b2 = *(const v16bf*)(wb2 + ko);
      v16bf b3 = *(const v16bf*)(wb3 + ko);
      acc0 = wmma_bf16(a, b0, acc0);
      acc1 = wmma_bf16(a, b1, acc1);
      acc2 = wmma_bf16(a, b2, acc2);
      acc3 = wmma_bf16(a, b3, acc3);
    }
  }

#define MPOLE_FIN_EPI(ACC, N)                                                  \
  {                                                                            \
    int   oc = (N) * 16 + col;                                                 \
    float bv = bc[oc];                                                         \
    float* orow = out + (((size_t)(b * 64 + oc)) * H + h) * W + w0 + khalf * 8;\
    _Pragma("unroll")                                                          \
    for (int r = 0; r < 8; ++r) orow[r] = ACC[r] + bv;                         \
  }
  MPOLE_FIN_EPI(acc0, 0)
  MPOLE_FIN_EPI(acc1, 1)
  MPOLE_FIN_EPI(acc2, 2)
  MPOLE_FIN_EPI(acc3, 3)
#undef MPOLE_FIN_EPI
}

// ---------------------------------------------------------------------------
// Launch
// ---------------------------------------------------------------------------
extern "C" void kernel_launch(void* const* d_in, const int* in_sizes, int n_in,
                              void* d_out, int out_size, void* d_ws, size_t ws_size,
                              hipStream_t stream) {
  const float* x  = (const float*)d_in[0];   // (4,64,256,256)
  const float* Ws = (const float*)d_in[1];   // (4,64,64,3,3)
  const float* bs = (const float*)d_in[2];   // (4,64)
  const float* Wc = (const float*)d_in[3];   // (64,256,3,3)
  const float* bc = (const float*)d_in[4];   // (64,)
  float* out = (float*)d_out;

  char* ws = (char*)d_ws;
  __bf16* L0  = (__bf16*)(ws);                         // 33,554,432 B
  __bf16* L1  = (__bf16*)(ws + 33554432);              //  8,388,608 B
  __bf16* L2  = (__bf16*)(ws + 41943040);              //  2,097,152 B
  __bf16* L3  = (__bf16*)(ws + 44040192);              //    524,288 B
  __bf16* WL  = (__bf16*)(ws + 44564480);              //    294,912 B
  __bf16* WF  = (__bf16*)(ws + 44859392);              //    294,912 B
  __bf16* CAT = (__bf16*)(ws + 45154304);              // 134,217,728 B

  // Stage 1: pyramid levels in NHWC bf16
  mpole_x_to_nhwc<<<4096, 256, 0, stream>>>(x, L0);
  mpole_pool_nhwc_bf16<<<(4 * 128 * 128 * 64) / 256, 256, 0, stream>>>(x, L1, 128, 128, 2);
  mpole_pool_nhwc_bf16<<<(4 *  64 *  64 * 64) / 256, 256, 0, stream>>>(x, L2,  64,  64, 4);
  mpole_pool_nhwc_bf16<<<(4 *  32 *  32 * 64) / 256, 256, 0, stream>>>(x, L3,  32,  32, 8);

  // Stage 2: weight reorder + bf16 convert
  mpole_reorder_wlvl<<<147456 / 256, 256, 0, stream>>>(Ws, WL);
  mpole_reorder_wfin<<<147456 / 256, 256, 0, stream>>>(Wc, WF);

  // Stage 3: level convs (dilated-conv-on-upsample == conv-at-level + upsample)
  mpole_conv_level<<<(4 * 256 * 256) / 64, 128, 0, stream>>>(L0, WL + 0 * 64 * 576, bs + 0 * 64, CAT, 256, 256, 0);
  mpole_conv_level<<<(4 * 128 * 128) / 64, 128, 0, stream>>>(L1, WL + 1 * 64 * 576, bs + 1 * 64, CAT, 128, 128, 1);
  mpole_conv_level<<<(4 *  64 *  64) / 64, 128, 0, stream>>>(L2, WL + 2 * 64 * 576, bs + 2 * 64, CAT,  64,  64, 2);
  mpole_conv_level<<<(4 *  32 *  32) / 64, 128, 0, stream>>>(L3, WL + 3 * 64 * 576, bs + 3 * 64, CAT,  32,  32, 3);

  // Stage 4: final 256->64 conv, NCHW f32 output
  mpole_conv_final<<<(4 * 256 * 256) / 64, 128, 0, stream>>>(CAT, WF, bc, out);
}
